// Sub_lstm_layer_35777077576481
// MI455X (gfx1250) — compile-verified, matrix operand order checked by reading, not run on hardware
//
#include <hip/hip_runtime.h>
#include <hip/hip_bf16.h>

// LSTM: B=64, T=512, IN=512, H=1024, gates 4H=4096.
// f16 WMMA (v_wmma_f32_16x16x32_f16) with f32 accumulation.

#define BB   64
#define TT   512
#define IN_  512
#define HH   1024
#define G4H  4096

typedef __attribute__((ext_vector_type(16))) _Float16 v16h;
typedef __attribute__((ext_vector_type(8)))  _Float16 v8h;
typedef __attribute__((ext_vector_type(8)))  float    v8f;

// ---------------------------------------------------------------------------
// Fragment loader for 16-bit A (and B == A-layout of B^T) per CDNA5 ISA 7.12.2:
// lane L: row = row0 + (L%16), half = L/16,
// elements 0..7  = K[k0 + 8*half      .. +7]  (contiguous 16 bytes)
// elements 8..15 = K[k0 + 16 + 8*half .. +7]  (contiguous 16 bytes)
// ---------------------------------------------------------------------------
__device__ __forceinline__ v16h load_frag16(const _Float16* __restrict__ M,
                                            int ld, int row0, int k0, int lane) {
    int r    = lane & 15;
    int half = lane >> 4;
    const _Float16* p = M + (size_t)(row0 + r) * (size_t)ld + (k0 + 8 * half);
    v8h lo = *(const v8h*)(p);        // global_load_b128
    v8h hi = *(const v8h*)(p + 16);
    return __builtin_shufflevector(lo, hi, 0,1,2,3,4,5,6,7,8,9,10,11,12,13,14,15);
}

__device__ __forceinline__ v8f wmma_f16(v16h a, v16h b, v8f c) {
    return __builtin_amdgcn_wmma_f32_16x16x32_f16(
        /*neg_a=*/false, a, /*neg_b=*/false, b,
        /*c_mod=*/(short)0, c, /*reuse_a=*/false, /*reuse_b=*/false);
}

__device__ __forceinline__ float sigf(float x) {
    return 1.0f / (1.0f + __expf(-x));
}

// ---------------------------------------------------------------------------
// f32 -> f16 elementwise convert
// ---------------------------------------------------------------------------
__global__ void __launch_bounds__(256) f32_to_f16_kernel(const float* __restrict__ s,
                                                         _Float16* __restrict__ d, int n) {
    int i = blockIdx.x * 256 + threadIdx.x;
    if (i < n) d[i] = (_Float16)s[i];
}

// ---------------------------------------------------------------------------
// x_proj[m, n] = sum_k X[m,k] * W_ih[n,k] + (b_ih[n] + b_hh[n]), stored f16.
// M = B*T = 32768 (m = b*T + t), N = 4H = 4096, K = IN = 512.
// One wave -> 32x64 strip (2 M-tiles x 4 N-tiles). All 6 fragments of a
// k-step are live simultaneously so their 12 b128 loads clause together and
// the 8 WMMAs drain the load counter progressively (no full-drain stalls).
// Grid: 16384 blocks x 128 threads = 65536 waves = 1024 Mpairs * 64 strips.
// ---------------------------------------------------------------------------
__global__ void __launch_bounds__(128) xproj_gemm_kernel(
    const _Float16* __restrict__ X,     // [32768, 512]
    const _Float16* __restrict__ Wih,   // [4096, 512]
    const float* __restrict__ b_ih,     // [4096]
    const float* __restrict__ b_hh,     // [4096]
    _Float16* __restrict__ XP)          // [32768, 4096]
{
    int lane = threadIdx.x & 31;
    int wave = threadIdx.x >> 5;
    int gw   = blockIdx.x * 4 + wave;      // 0 .. 65535
    int mPair  = gw >> 6;                  // 0 .. 1023  (32 rows each)
    int nStrip = gw & 63;                  // 0 .. 63    (64 cols each)
    int m0 = mPair * 32;
    int n0 = nStrip * 64;

    v8f acc[2][4] = {};
    for (int k0 = 0; k0 < IN_; k0 += 32) {
        // Batch all fragment loads for this k-step (independent dest regs).
        v16h a0 = load_frag16(X, IN_, m0,      k0, lane);
        v16h a1 = load_frag16(X, IN_, m0 + 16, k0, lane);
        v16h bf0 = load_frag16(Wih, IN_, n0 +  0, k0, lane);
        v16h bf1 = load_frag16(Wih, IN_, n0 + 16, k0, lane);
        v16h bf2 = load_frag16(Wih, IN_, n0 + 32, k0, lane);
        v16h bf3 = load_frag16(Wih, IN_, n0 + 48, k0, lane);
        acc[0][0] = wmma_f16(a0, bf0, acc[0][0]);
        acc[1][0] = wmma_f16(a1, bf0, acc[1][0]);
        acc[0][1] = wmma_f16(a0, bf1, acc[0][1]);
        acc[1][1] = wmma_f16(a1, bf1, acc[1][1]);
        acc[0][2] = wmma_f16(a0, bf2, acc[0][2]);
        acc[1][2] = wmma_f16(a1, bf2, acc[1][2]);
        acc[0][3] = wmma_f16(a0, bf3, acc[0][3]);
        acc[1][3] = wmma_f16(a1, bf3, acc[1][3]);
    }

    // Epilogue: C layout -> n = n0 + 16j + (lane&15); m = v + 8*(lane>>4).
    int r = lane & 15, half = lane >> 4;
    #pragma unroll
    for (int j = 0; j < 4; ++j) {
        int n = n0 + 16 * j + r;
        float bias = b_ih[n] + b_hh[n];
        #pragma unroll
        for (int mt = 0; mt < 2; ++mt) {
            #pragma unroll
            for (int v = 0; v < 8; ++v) {
                int m = m0 + 16 * mt + v + 8 * half;
                XP[(size_t)m * G4H + n] = (_Float16)(acc[mt][j][v] + bias);
            }
        }
    }
}

// ---------------------------------------------------------------------------
// One LSTM timestep: gates = xp_t + h_in @ W_hh^T ; pointwise update.
// Block = 4 waves, one (b-tile 16 x h-tile 16), split-K: wave w does
// K in [w*256, w*256+256), 8 k-steps, double-buffered fragments.
// LDS reduction of partial gate tiles, then fused i/f/g/o update.
// Grid: 256 blocks x 128 threads = 1024 waves = (4 b-tiles * 64 h-tiles) * 4.
// ---------------------------------------------------------------------------
__global__ void __launch_bounds__(128) lstm_step_kernel(
    const _Float16* __restrict__ XP,     // [32768, 4096]
    const _Float16* __restrict__ Whh,    // [4096, 1024]
    const _Float16* __restrict__ h_in,   // [64, 1024]
    _Float16* __restrict__ h_out,        // [64, 1024]
    float* __restrict__ c,               // [64, 1024]
    float* __restrict__ h_fin,           // [64, 1024] (d_out)
    int t)
{
    __shared__ float part[4][4][32][8];  // [wave][gate][lane][vgpr] = 16 KB

    int lane = threadIdx.x & 31;
    int wave = threadIdx.x >> 5;
    int mTile = blockIdx.x >> 6;         // 0 .. 3
    int hTile = blockIdx.x & 63;         // 0 .. 63
    int m0 = mTile * 16;
    int h0 = hTile * 16;
    int r = lane & 15, half = lane >> 4;
    int kbase = wave * (HH / 4);         // 256-wide K slice per wave

    // ---- Phase 1: partial GEMM over this wave's K slice (double-buffered) ----
    v8f acc[4] = {};
    v16h a = load_frag16(h_in, HH, m0, kbase, lane);
    v16h b0 = load_frag16(Whh, HH, 0 * HH + h0, kbase, lane);
    v16h b1 = load_frag16(Whh, HH, 1 * HH + h0, kbase, lane);
    v16h b2 = load_frag16(Whh, HH, 2 * HH + h0, kbase, lane);
    v16h b3 = load_frag16(Whh, HH, 3 * HH + h0, kbase, lane);
    #pragma unroll
    for (int i = 0; i < 8; ++i) {
        v16h an, c0, c1, c2, c3;
        if (i < 7) {                     // prefetch next K chunk before WMMAs
            int kn = kbase + (i + 1) * 32;
            an = load_frag16(h_in, HH, m0, kn, lane);
            c0 = load_frag16(Whh, HH, 0 * HH + h0, kn, lane);
            c1 = load_frag16(Whh, HH, 1 * HH + h0, kn, lane);
            c2 = load_frag16(Whh, HH, 2 * HH + h0, kn, lane);
            c3 = load_frag16(Whh, HH, 3 * HH + h0, kn, lane);
        }
        acc[0] = wmma_f16(a, b0, acc[0]);
        acc[1] = wmma_f16(a, b1, acc[1]);
        acc[2] = wmma_f16(a, b2, acc[2]);
        acc[3] = wmma_f16(a, b3, acc[3]);
        if (i < 7) { a = an; b0 = c0; b1 = c1; b2 = c2; b3 = c3; }
    }

    #pragma unroll
    for (int g = 0; g < 4; ++g)
        #pragma unroll
        for (int v = 0; v < 8; ++v)
            part[wave][g][lane][v] = acc[g][v];
    __syncthreads();

    // ---- Phase 2: wave g reduces gate g, adds x_proj, applies nonlinearity ----
    {
        int g = wave;
        int n = g * HH + h0 + r;
        float red[8];
        #pragma unroll
        for (int v = 0; v < 8; ++v) {
            red[v] = part[0][g][lane][v] + part[1][g][lane][v] +
                     part[2][g][lane][v] + part[3][g][lane][v];
            int bRow = m0 + v + 8 * half;
            red[v] += (float)XP[((size_t)bRow * TT + t) * G4H + n];
            red[v] = (g == 2) ? tanhf(red[v]) : sigf(red[v]);
            part[0][g][lane][v] = red[v];
        }
    }
    __syncthreads();

    // ---- Phase 3: fused cell/hidden update; wave w handles v = {2w, 2w+1} ----
    #pragma unroll
    for (int j = 0; j < 2; ++j) {
        int v = wave * 2 + j;
        int bRow = m0 + v + 8 * half;
        int col  = h0 + r;
        size_t idx = (size_t)bRow * HH + col;
        float ig = part[0][0][lane][v];
        float fg = part[0][1][lane][v];
        float gg = part[0][2][lane][v];
        float og = part[0][3][lane][v];
        float cn = fg * c[idx] + ig * gg;
        c[idx] = cn;
        float hn = og * tanhf(cn);
        h_out[idx] = (_Float16)hn;
        if (t == TT - 1) h_fin[idx] = hn;
    }
}

// ---------------------------------------------------------------------------
// Host launcher. Inputs: input[B,T,IN], W_ih[4H,IN], W_hh[4H,H], b_ih[4H], b_hh[4H].
// Workspace layout (bytes):
//   Xf16   : 32768*512*2   =  33554432
//   Wih16  : 4096*512*2    =   4194304
//   Whh16  : 4096*1024*2   =   8388608
//   XP16   : 32768*4096*2  = 268435456
//   h16[2] : 2*64*1024*2   =    262144
//   c32    : 64*1024*4     =    262144
// ---------------------------------------------------------------------------
extern "C" void kernel_launch(void* const* d_in, const int* in_sizes, int n_in,
                              void* d_out, int out_size, void* d_ws, size_t ws_size,
                              hipStream_t stream) {
    const float* input = (const float*)d_in[0];
    const float* W_ih  = (const float*)d_in[1];
    const float* W_hh  = (const float*)d_in[2];
    const float* b_ih  = (const float*)d_in[3];
    const float* b_hh  = (const float*)d_in[4];

    char* ws = (char*)d_ws;
    _Float16* Xf16  = (_Float16*)ws;  ws += (size_t)BB * TT * IN_ * 2;
    _Float16* Wih16 = (_Float16*)ws;  ws += (size_t)G4H * IN_ * 2;
    _Float16* Whh16 = (_Float16*)ws;  ws += (size_t)G4H * HH * 2;
    _Float16* XP16  = (_Float16*)ws;  ws += (size_t)BB * TT * G4H * 2;
    _Float16* h16a  = (_Float16*)ws;  ws += (size_t)BB * HH * 2;
    _Float16* h16b  = (_Float16*)ws;  ws += (size_t)BB * HH * 2;
    float*    c32   = (float*)ws;     ws += (size_t)BB * HH * 4;

    // f32 -> f16 conversions
    {
        int n = BB * TT * IN_;
        f32_to_f16_kernel<<<(n + 255) / 256, 256, 0, stream>>>(input, Xf16, n);
        n = G4H * IN_;
        f32_to_f16_kernel<<<(n + 255) / 256, 256, 0, stream>>>(W_ih, Wih16, n);
        n = G4H * HH;
        f32_to_f16_kernel<<<(n + 255) / 256, 256, 0, stream>>>(W_hh, Whh16, n);
    }

    // h0 = 0, c0 = 0
    (void)hipMemsetAsync(h16a, 0, (size_t)BB * HH * 2, stream);
    (void)hipMemsetAsync(c32,  0, (size_t)BB * HH * 4, stream);

    // x_proj = X @ W_ih^T + (b_ih + b_hh)
    xproj_gemm_kernel<<<16384, 128, 0, stream>>>(Xf16, Wih16, b_ih, b_hh, XP16);

    // Sequential recurrence; ping-pong h buffers (stream order = step order).
    for (int t = 0; t < TT; ++t) {
        const _Float16* hin  = (t & 1) ? h16b : h16a;
        _Float16*       hout = (t & 1) ? h16a : h16b;
        lstm_step_kernel<<<256, 128, 0, stream>>>(XP16, Whh16, hin, hout, c32,
                                                  (float*)d_out, t);
    }
}
